// MambaWrapper_74998718922874
// MI455X (gfx1250) — compile-verified, matrix operand order checked by reading, not run on hardware
//
#include <hip/hip_runtime.h>
#include <hip/hip_bf16.h>

// ---------------------------------------------------------------------------
// Mamba-2 block forward for MI455X (gfx1250, wave32, WMMA).
// All heavy GEMMs + SSD chunk matmuls use v_wmma_f32_16x16x32_bf16.
// GEMM B-operand K=128 panels are staged into LDS with
// global_load_async_to_lds_b128 (ASYNCcnt), double-buffered, one barrier per
// 4 K-steps so the inner WMMA loop stays fully unrolled/pipelined.
// Large intermediates are carried as bf16 to respect the HBM roofline.
// ---------------------------------------------------------------------------

#define B_SZ     8
#define LSEQ     2048
#define D_MODEL  1024
#define D_INNER  2048
#define D_STATE  128
#define NHEADS   32
#define HEADDIM  64
#define CONV_DIM 2304          // D_INNER + 2*D_STATE
#define D_PROJ   4384          // 2*D_INNER + 2*D_STATE + NHEADS
#define NCHUNK   32
#define CHUNKL   64
#define MROWS    (B_SZ * LSEQ) // 16384
#define KPANEL   128           // K-panel staged per sync

typedef __attribute__((ext_vector_type(16))) __bf16 v16bf;
typedef __attribute__((ext_vector_type(8)))  __bf16 v8bf;
typedef __attribute__((ext_vector_type(8)))  float  v8f;

static_assert(sizeof(__bf16) == 2, "bf16 size");

// --- WMMA fragment helpers (bf16 16x16x32, wave32 layouts per CDNA5 ISA) ---

__device__ __forceinline__ v16bf ld_frag(const __bf16* base, int ld, int row_base, int k_base) {
    int lane = threadIdx.x & 31;
    const __bf16* p = base + (size_t)(row_base + (lane & 15)) * ld
                           + (size_t)(k_base + ((lane >> 4) << 3));
    v8bf lo = *(const v8bf*)p;
    v8bf hi = *(const v8bf*)(p + 16);
    v16bf f;
#pragma unroll
    for (int i = 0; i < 8; ++i) { f[i] = lo[i]; f[8 + i] = hi[i]; }
    return f;
}

__device__ __forceinline__ v8f wmma_bf(v16bf a, v16bf b, v8f c) {
    return __builtin_amdgcn_wmma_f32_16x16x32_bf16(false, a, false, b, (short)0, c, false, false);
}

__device__ __forceinline__ int frag_row(int r) { return (((threadIdx.x & 31) >> 4) << 3) + r; }
__device__ __forceinline__ int frag_col()      { return threadIdx.x & 15; }

// --- async stage of one 32(N) x 128(K) bf16 panel (8KB) into LDS -----------
// Issued by one wave (32 lanes x 16B x 16 issues). Tracked by ASYNCcnt.
// LDS layout: [n][k], row stride 128 elements (256B).

__device__ __forceinline__ void async_stage_panel(const __bf16* Wpanel, int K,
                                                  unsigned lds_base, int lane) {
#pragma unroll
    for (int j = 0; j < 16; ++j) {
        int byteoff = j * 512 + lane * 16;      // 0..8191, 16B per lane
        int n  = byteoff >> 8;                  // row within panel (256B rows)
        int kb = (byteoff & 255) >> 1;          // bf16 element within row
        unsigned long long g = (unsigned long long)(uintptr_t)(Wpanel + (size_t)n * K + kb);
        unsigned l = lds_base + (unsigned)byteoff;
        asm volatile("global_load_async_to_lds_b128 %0, %1, off"
                     :: "v"(l), "v"(g) : "memory");
    }
}

__device__ __forceinline__ void wait_asynccnt0() {
    asm volatile("s_wait_asynccnt 0x0" ::: "memory");
}

// --- elementwise f32 -> bf16 convert ---------------------------------------

__global__ void __launch_bounds__(256) cvt_bf16_kernel(const float* __restrict__ in,
                                                       __bf16* __restrict__ out, long n) {
    long i = (long)blockIdx.x * 256 + threadIdx.x;
    if (i < n) out[i] = (__bf16)in[i];
}

// --- tiled GEMM: C[M,N] = A[M,K] @ W[N,K]^T, bf16 inputs, f32 or bf16 out --
// block = 256 threads = 8 waves; wave -> 32 rows; block tile 256x32.
// W K-panels (shared by all waves) async-staged into LDS, double buffered.

template <bool BF16OUT>
__global__ void __launch_bounds__(256) gemm_nt_kernel(const __bf16* __restrict__ A,
                                                      const __bf16* __restrict__ W,
                                                      float* __restrict__ Cf,
                                                      __bf16* __restrict__ Cb,
                                                      int N, int K) {
    __shared__ __bf16 sW[2][32 * KPANEL];       // double-buffered 8KB W panels
    int wid  = threadIdx.x >> 5;
    int lane = threadIdx.x & 31;
    int m0 = blockIdx.x * 256 + wid * 32;
    int n0 = blockIdx.y * 32;

    if (wid == 0)
        async_stage_panel(W + (size_t)n0 * K, K, (unsigned)(uintptr_t)&sW[0][0], lane);

    v8f a00 = {}, a01 = {}, a10 = {}, a11 = {};
    for (int kp = 0, it = 0; kp < K; kp += KPANEL, ++it) {
        int buf = it & 1;
        if (wid == 0) wait_asynccnt0();         // panel for this step has landed
        __syncthreads();                        // publish LDS panel to all waves
        if (wid == 0 && kp + KPANEL < K)        // kick off next panel copy
            async_stage_panel(W + (size_t)n0 * K + (kp + KPANEL), K,
                              (unsigned)(uintptr_t)&sW[buf ^ 1][0], lane);
        if (kp + KPANEL < K)
            __builtin_prefetch(A + (size_t)m0 * K + kp + KPANEL, 0, 3);

#pragma unroll
        for (int kk = 0; kk < KPANEL; kk += 32) {
            v16bf fa0 = ld_frag(A, K, m0,      kp + kk);
            v16bf fa1 = ld_frag(A, K, m0 + 16, kp + kk);
            v16bf fb0 = ld_frag(sW[buf], KPANEL, 0,  kk);
            v16bf fb1 = ld_frag(sW[buf], KPANEL, 16, kk);
            a00 = wmma_bf(fa0, fb0, a00);
            a01 = wmma_bf(fa0, fb1, a01);
            a10 = wmma_bf(fa1, fb0, a10);
            a11 = wmma_bf(fa1, fb1, a11);
        }
    }
    int col = frag_col();
#pragma unroll
    for (int r = 0; r < 8; ++r) {
        int rr = frag_row(r);
        size_t i00 = (size_t)(m0 + rr)      * N + n0 + col;
        size_t i01 = (size_t)(m0 + rr)      * N + n0 + 16 + col;
        size_t i10 = (size_t)(m0 + 16 + rr) * N + n0 + col;
        size_t i11 = (size_t)(m0 + 16 + rr) * N + n0 + 16 + col;
        if constexpr (BF16OUT) {
            Cb[i00] = (__bf16)a00[r]; Cb[i01] = (__bf16)a01[r];
            Cb[i10] = (__bf16)a10[r]; Cb[i11] = (__bf16)a11[r];
        } else {
            Cf[i00] = a00[r]; Cf[i01] = a01[r];
            Cf[i10] = a10[r]; Cf[i11] = a11[r];
        }
    }
}

// --- LayerNorm (adds Wi bias, writes bf16), one block per row of 1024 ------

__global__ void __launch_bounds__(256) layernorm_kernel(const float* __restrict__ U,
                                                        const float* __restrict__ bi,
                                                        const float* __restrict__ g,
                                                        const float* __restrict__ bb,
                                                        __bf16* __restrict__ O) {
    int row = blockIdx.x;
    const float* u = U + (size_t)row * D_MODEL;
    float v[4], lsum = 0.f, lsq = 0.f;
#pragma unroll
    for (int i = 0; i < 4; ++i) {
        int ch = threadIdx.x + i * 256;
        float x = u[ch] + bi[ch];
        v[i] = x; lsum += x; lsq += x * x;
    }
    __shared__ float red[512];
    red[threadIdx.x] = lsum; red[256 + threadIdx.x] = lsq;
    __syncthreads();
    for (int s = 128; s > 0; s >>= 1) {
        if (threadIdx.x < s) {
            red[threadIdx.x]       += red[threadIdx.x + s];
            red[256 + threadIdx.x] += red[256 + threadIdx.x + s];
        }
        __syncthreads();
    }
    float mu  = red[0] * (1.f / D_MODEL);
    float var = red[256] * (1.f / D_MODEL) - mu * mu;
    float rs  = rsqrtf(var + 1e-5f);
#pragma unroll
    for (int i = 0; i < 4; ++i) {
        int ch = threadIdx.x + i * 256;
        O[(size_t)row * D_MODEL + ch] = (__bf16)((v[i] - mu) * rs * g[ch] + bb[ch]);
    }
}

// --- dt = softplus(zx[..., 4352+h] + dt_bias[h]) ---------------------------

__global__ void __launch_bounds__(256) dt_kernel(const __bf16* __restrict__ ZX,
                                                 const float* __restrict__ dtb,
                                                 float* __restrict__ dt, int total) {
    int idx = blockIdx.x * 256 + threadIdx.x;
    if (idx >= total) return;
    int h = idx & 31;
    size_t row = (size_t)(idx >> 5);
    float x = (float)ZX[row * D_PROJ + (D_INNER + CONV_DIM) + h] + dtb[h];
    dt[idx] = (x > 20.f) ? x : log1pf(expf(x));
}

// --- causal depthwise conv + SiLU + split into xs / x*dt / B / C -----------

__global__ void __launch_bounds__(256) convact_kernel(const __bf16* __restrict__ ZX,
                                                      const float* __restrict__ cw,
                                                      const float* __restrict__ cb,
                                                      const float* __restrict__ dt,
                                                      __bf16* __restrict__ xs,
                                                      __bf16* __restrict__ xdt,
                                                      __bf16* __restrict__ Bo,
                                                      __bf16* __restrict__ Co, long total) {
    long idx = (long)blockIdx.x * 256 + threadIdx.x;
    if (idx >= total) return;
    int  ch = (int)(idx % CONV_DIM);
    long bl = idx / CONV_DIM;
    int  l  = (int)(bl % LSEQ);
    long b  = bl / LSEQ;
    float acc = cb[ch];
#pragma unroll
    for (int k = 0; k < 4; ++k) {
        int lp = l + k - 3;
        if (lp >= 0)
            acc += (float)ZX[((size_t)b * LSEQ + lp) * D_PROJ + D_INNER + ch] * cw[ch * 4 + k];
    }
    float s = acc / (1.f + expf(-acc));   // SiLU
    size_t row = (size_t)b * LSEQ + l;
    if (ch < D_INNER) {
        int h = ch >> 6;
        xs [row * D_INNER + ch] = (__bf16)s;
        xdt[row * D_INNER + ch] = (__bf16)(s * dt[row * NHEADS + h]);
    } else if (ch < D_INNER + D_STATE) {
        Bo[row * D_STATE + (ch - D_INNER)] = (__bf16)s;
    } else {
        Co[row * D_STATE + (ch - D_INNER - D_STATE)] = (__bf16)s;
    }
}

// --- per-chunk cumsum of A*dt: Acs[b][h][c][j] -----------------------------

__global__ void __launch_bounds__(256) acs_kernel(const float* __restrict__ dt,
                                                  const float* __restrict__ A_log,
                                                  float* __restrict__ Acs) {
    int idx = blockIdx.x * 256 + threadIdx.x;       // 8192 = b*32h*32c
    int c = idx & 31, h = (idx >> 5) & 31, b = idx >> 10;
    float A = -expf(A_log[h]);
    float run = 0.f;
    float* o = Acs + (size_t)idx * CHUNKL;
    for (int j = 0; j < CHUNKL; ++j) {
        run += A * dt[((size_t)b * LSEQ + c * CHUNKL + j) * NHEADS + h];
        o[j] = run;
    }
}

// --- SSD per-chunk: CB, masked Y_diag, local states (all WMMA) -------------
// one block (256 thr / 8 waves) per (b,c,h); blockIdx = ((b*32+c)*32+h)

__global__ void __launch_bounds__(256) ssd_chunk_kernel(const __bf16* __restrict__ Bm,
                                                        const __bf16* __restrict__ Cm,
                                                        const __bf16* __restrict__ Xdt,
                                                        const float* __restrict__ Acs,
                                                        __bf16* __restrict__ Sloc,
                                                        float* __restrict__ Y) {
    int h = blockIdx.x & 31, c = (blockIdx.x >> 5) & 31, b = blockIdx.x >> 10;
    __shared__ float  sA[CHUNKL];
    __shared__ __bf16 sG [CHUNKL * CHUNKL];   // masked CB, row l, s contiguous
    __shared__ __bf16 sXT[CHUNKL * HEADDIM];  // XT[p][l], l contiguous
    __shared__ __bf16 sBT[D_STATE * CHUNKL];  // BT[n][l] * decay_states, l contiguous

    const float* acs = Acs + (size_t)((b * 32 + h) * 32 + c) * CHUNKL;
    size_t rowBL = (size_t)b * LSEQ + c * CHUNKL;
    if (threadIdx.x < CHUNKL) sA[threadIdx.x] = acs[threadIdx.x];
    __syncthreads();
    float alast = sA[CHUNKL - 1];

    for (int idx = threadIdx.x; idx < CHUNKL * HEADDIM; idx += 256) {
        int l = idx >> 6, p = idx & 63;
        sXT[p * CHUNKL + l] = Xdt[(rowBL + l) * D_INNER + h * HEADDIM + p];
    }
    for (int idx = threadIdx.x; idx < CHUNKL * D_STATE; idx += 256) {
        int l = idx >> 7, n = idx & 127;
        float ds = expf(alast - sA[l]);
        sBT[n * CHUNKL + l] = (__bf16)((float)Bm[(rowBL + l) * D_STATE + n] * ds);
    }
    __syncthreads();

    int wid = threadIdx.x >> 5, col = frag_col();
    const __bf16* Cg = Cm + rowBL * D_STATE;
    const __bf16* Bg = Bm + rowBL * D_STATE;

    // 1) G = mask(C @ B^T) * exp(segsum)
#pragma unroll
    for (int t = 0; t < 2; ++t) {
        int ti = wid * 2 + t, tl = ti >> 2, ts = ti & 3;
        v8f acc = {};
#pragma unroll
        for (int kk = 0; kk < D_STATE; kk += 32)
            acc = wmma_bf(ld_frag(Cg, D_STATE, tl * 16, kk),
                          ld_frag(Bg, D_STATE, ts * 16, kk), acc);
#pragma unroll
        for (int r = 0; r < 8; ++r) {
            int l = tl * 16 + frag_row(r), s = ts * 16 + col;
            float v = (l >= s) ? acc[r] * expf(sA[l] - sA[s]) : 0.f;
            sG[l * CHUNKL + s] = (__bf16)v;
        }
    }
    __syncthreads();

    // 2) Y_diag = G @ Xdt   (B-operand columns p -> sXT rows p)
#pragma unroll
    for (int t = 0; t < 2; ++t) {
        int ti = wid * 2 + t, tl = ti >> 2, tp = ti & 3;
        v8f acc = {};
#pragma unroll
        for (int kk = 0; kk < CHUNKL; kk += 32)
            acc = wmma_bf(ld_frag(sG,  CHUNKL, tl * 16, kk),
                          ld_frag(sXT, CHUNKL, tp * 16, kk), acc);
#pragma unroll
        for (int r = 0; r < 8; ++r) {
            int l = tl * 16 + frag_row(r), p = tp * 16 + col;
            Y[(rowBL + l) * D_INNER + h * HEADDIM + p] = acc[r];
        }
    }

    // 3) local states S[p][n] = sum_l Xdt[l,p] * (B[l,n]*decay)
    size_t sbase = (size_t)blockIdx.x * (HEADDIM * D_STATE);
#pragma unroll
    for (int t = 0; t < 4; ++t) {
        int ti = wid * 4 + t, tp = ti >> 3, tn = ti & 7;
        v8f acc = {};
#pragma unroll
        for (int kk = 0; kk < CHUNKL; kk += 32)
            acc = wmma_bf(ld_frag(sXT, CHUNKL, tp * 16, kk),
                          ld_frag(sBT, CHUNKL, tn * 16, kk), acc);
#pragma unroll
        for (int r = 0; r < 8; ++r) {
            int p = tp * 16 + frag_row(r), n = tn * 16 + col;
            Sloc[sbase + (size_t)p * D_STATE + n] = (__bf16)acc[r];
        }
    }
}

// --- inter-chunk scan: rewrite S in place (local -> prefix states) ---------

__global__ void __launch_bounds__(256) scan_kernel(__bf16* __restrict__ S,
                                                   const float* __restrict__ Acs) {
    int idx = blockIdx.x * 256 + threadIdx.x;      // 2,097,152 = b*h*p*n
    int n = idx & 127, p = (idx >> 7) & 63, h = (idx >> 13) & 31, b = idx >> 18;
    float run = 0.f;
    size_t pn = (size_t)p * D_STATE + n;
    for (int c = 0; c < NCHUNK; ++c) {
        size_t off = (size_t)((b * 32 + c) * 32 + h) * (HEADDIM * D_STATE) + pn;
        float loc = (float)S[off];
        S[off] = (__bf16)run;
        run = expf(Acs[(size_t)((b * 32 + h) * 32 + c) * CHUNKL + (CHUNKL - 1)]) * run + loc;
    }
}

// --- Y_off = out_decay * (C @ states^T) + D*xs, added into Y ---------------

__global__ void __launch_bounds__(256) ssd_yoff_kernel(const __bf16* __restrict__ Cm,
                                                       const __bf16* __restrict__ S,
                                                       const __bf16* __restrict__ xs,
                                                       const float* __restrict__ Acs,
                                                       const float* __restrict__ Dp,
                                                       float* __restrict__ Y) {
    int h = blockIdx.x & 31, c = (blockIdx.x >> 5) & 31, b = blockIdx.x >> 10;
    __shared__ float sA[CHUNKL];
    const float* acs = Acs + (size_t)((b * 32 + h) * 32 + c) * CHUNKL;
    if (threadIdx.x < CHUNKL) sA[threadIdx.x] = acs[threadIdx.x];
    __syncthreads();
    size_t rowBL = (size_t)b * LSEQ + c * CHUNKL;
    const __bf16* Cg = Cm + rowBL * D_STATE;
    const __bf16* Sg = S + (size_t)blockIdx.x * (HEADDIM * D_STATE);
    float Dh = Dp[h];
    int wid = threadIdx.x >> 5, col = frag_col();
#pragma unroll
    for (int t = 0; t < 2; ++t) {
        int ti = wid * 2 + t, tl = ti >> 2, tp = ti & 3;
        v8f acc = {};
#pragma unroll
        for (int kk = 0; kk < D_STATE; kk += 32)
            acc = wmma_bf(ld_frag(Cg, D_STATE, tl * 16, kk),
                          ld_frag(Sg, D_STATE, tp * 16, kk), acc);
#pragma unroll
        for (int r = 0; r < 8; ++r) {
            int l = tl * 16 + frag_row(r), p = tp * 16 + col;
            size_t yi = (rowBL + l) * D_INNER + h * HEADDIM + p;
            Y[yi] = Y[yi] + expf(sA[l]) * acc[r] + Dh * (float)xs[yi];
        }
    }
}

// --- gate with silu(z), RMSNorm over D_INNER, write bf16 -------------------

__global__ void __launch_bounds__(256) gate_rms_kernel(const float* __restrict__ Y,
                                                       const __bf16* __restrict__ ZX,
                                                       const float* __restrict__ w,
                                                       __bf16* __restrict__ O) {
    int row = blockIdx.x;
    float vals[8], sq = 0.f;
#pragma unroll
    for (int i = 0; i < 8; ++i) {
        int ch = threadIdx.x + i * 256;
        float y = Y[(size_t)row * D_INNER + ch];
        float z = (float)ZX[(size_t)row * D_PROJ + ch];
        float g = y * (z / (1.f + expf(-z)));
        vals[i] = g; sq += g * g;
    }
    __shared__ float red[256];
    red[threadIdx.x] = sq;
    __syncthreads();
    for (int s = 128; s > 0; s >>= 1) {
        if (threadIdx.x < s) red[threadIdx.x] += red[threadIdx.x + s];
        __syncthreads();
    }
    float scale = rsqrtf(red[0] * (1.f / D_INNER) + 1e-5f);
#pragma unroll
    for (int i = 0; i < 8; ++i) {
        int ch = threadIdx.x + i * 256;
        O[(size_t)row * D_INNER + ch] = (__bf16)(vals[i] * scale * w[ch]);
    }
}

// --- hidden = out[:, -1, :] ------------------------------------------------

__global__ void __launch_bounds__(256) hidden_kernel(float* __restrict__ out) {
    int idx = blockIdx.x * 256 + threadIdx.x;      // 8192
    int b = idx >> 10, d = idx & 1023;
    out[(size_t)MROWS * D_MODEL + idx] = out[((size_t)b * LSEQ + (LSEQ - 1)) * D_MODEL + d];
}

// ---------------------------------------------------------------------------

extern "C" void kernel_launch(void* const* d_in, const int* in_sizes, int n_in,
                              void* d_out, int out_size, void* d_ws, size_t ws_size,
                              hipStream_t stream) {
    (void)in_sizes; (void)n_in; (void)out_size; (void)ws_size;
    const float* x       = (const float*)d_in[0];
    const float* Wi      = (const float*)d_in[1];
    const float* bi      = (const float*)d_in[2];
    const float* ln_g    = (const float*)d_in[3];
    const float* ln_b    = (const float*)d_in[4];
    const float* Win     = (const float*)d_in[5];
    const float* conv_w  = (const float*)d_in[6];
    const float* conv_b  = (const float*)d_in[7];
    const float* dt_bias = (const float*)d_in[8];
    const float* A_log   = (const float*)d_in[9];
    const float* Dp      = (const float*)d_in[10];
    const float* rms_w   = (const float*)d_in[11];
    const float* Wout    = (const float*)d_in[12];
    float* out = (float*)d_out;

    char* ws = (char*)d_ws;
    size_t off = 0;
    auto alloc = [&](size_t bytes) -> void* {
        void* p = ws + off;
        off += (bytes + 255) & ~(size_t)255;
        return p;
    };

    __bf16* xbf    = (__bf16*)alloc((size_t)MROWS * D_MODEL * 2);
    __bf16* Wibf   = (__bf16*)alloc((size_t)D_MODEL * D_MODEL * 2);
    __bf16* Winbf  = (__bf16*)alloc((size_t)D_PROJ * D_MODEL * 2);
    __bf16* Woutbf = (__bf16*)alloc((size_t)D_MODEL * D_INNER * 2);
    float*  u      = (float*) alloc((size_t)MROWS * D_MODEL * 4);
    __bf16* ulnbf  = (__bf16*)alloc((size_t)MROWS * D_MODEL * 2);
    __bf16* zxbf   = (__bf16*)alloc((size_t)MROWS * D_PROJ * 2);
    __bf16* xsbf   = (__bf16*)alloc((size_t)MROWS * D_INNER * 2);
    __bf16* xdtbf  = (__bf16*)alloc((size_t)MROWS * D_INNER * 2);
    __bf16* Bbf    = (__bf16*)alloc((size_t)MROWS * D_STATE * 2);
    __bf16* Cbf    = (__bf16*)alloc((size_t)MROWS * D_STATE * 2);
    float*  dt     = (float*) alloc((size_t)MROWS * NHEADS * 4);
    float*  Acs    = (float*) alloc((size_t)8192 * CHUNKL * 4);
    __bf16* S      = (__bf16*)alloc((size_t)8192 * HEADDIM * D_STATE * 2);
    float*  Yb     = (float*) alloc((size_t)MROWS * D_INNER * 4);
    __bf16* ybf    = (__bf16*)alloc((size_t)MROWS * D_INNER * 2);

    // bf16 staging
    cvt_bf16_kernel<<<(MROWS * D_MODEL + 255) / 256, 256, 0, stream>>>(x, xbf, (long)MROWS * D_MODEL);
    cvt_bf16_kernel<<<(D_MODEL * D_MODEL + 255) / 256, 256, 0, stream>>>(Wi, Wibf, (long)D_MODEL * D_MODEL);
    cvt_bf16_kernel<<<(D_PROJ * D_MODEL + 255) / 256, 256, 0, stream>>>(Win, Winbf, (long)D_PROJ * D_MODEL);
    cvt_bf16_kernel<<<(D_MODEL * D_INNER + 255) / 256, 256, 0, stream>>>(Wout, Woutbf, (long)D_MODEL * D_INNER);

    // u = x @ Wi^T  (bias folded into LN)
    gemm_nt_kernel<false><<<dim3(MROWS / 256, D_MODEL / 32), 256, 0, stream>>>(xbf, Wibf, u, nullptr, D_MODEL, D_MODEL);
    layernorm_kernel<<<MROWS, 256, 0, stream>>>(u, bi, ln_g, ln_b, ulnbf);

    // zxbcdt = u_ln @ Win^T  (bf16 out)
    gemm_nt_kernel<true><<<dim3(MROWS / 256, D_PROJ / 32), 256, 0, stream>>>(ulnbf, Winbf, nullptr, zxbf, D_PROJ, D_MODEL);

    // dt, conv+silu+split, chunk cumsum
    dt_kernel<<<(MROWS * NHEADS + 255) / 256, 256, 0, stream>>>(zxbf, dt_bias, dt, MROWS * NHEADS);
    convact_kernel<<<(int)(((long)MROWS * CONV_DIM + 255) / 256), 256, 0, stream>>>(
        zxbf, conv_w, conv_b, dt, xsbf, xdtbf, Bbf, Cbf, (long)MROWS * CONV_DIM);
    acs_kernel<<<32, 256, 0, stream>>>(dt, A_log, Acs);

    // SSD: per-chunk WMMA, inter-chunk scan, off-diagonal WMMA
    ssd_chunk_kernel<<<8192, 256, 0, stream>>>(Bbf, Cbf, xdtbf, Acs, S, Yb);
    scan_kernel<<<8192, 256, 0, stream>>>(S, Acs);
    ssd_yoff_kernel<<<8192, 256, 0, stream>>>(Cbf, S, xsbf, Acs, Dp, Yb);

    // gate + RMSNorm, out_proj, hidden
    gate_rms_kernel<<<MROWS, 256, 0, stream>>>(Yb, zxbf, rms_w, ybf);
    gemm_nt_kernel<false><<<dim3(MROWS / 256, D_MODEL / 32), 256, 0, stream>>>(ybf, Woutbf, out, nullptr, D_MODEL, D_INNER);
    hidden_kernel<<<32, 256, 0, stream>>>(out);
}
